// ShortestPathPredictor_82884278878925
// MI455X (gfx1250) — matrix operand order repaired and checked by reading, not compiled, not run on hardware
//
#include <hip/hip_runtime.h>

typedef __attribute__((ext_vector_type(16))) _Float16 v16h;
typedef __attribute__((ext_vector_type(8)))  _Float16 v8h;
typedef __attribute__((ext_vector_type(4)))  _Float16 v4h;
typedef __attribute__((ext_vector_type(8)))  float    v8f;
typedef __attribute__((ext_vector_type(4)))  float    v4f;

constexpr int HID  = 128;   // hidden size
constexpr int KDIM = 256;   // 2*HID (concat)
constexpr int MT   = 64;    // edges per block
constexpr int LDA  = 264;   // sA row stride (halves): 528B -> stride 4 banks, conflict-free b128
constexpr int LDH  = 136;   // sH row stride (halves): 272B -> stride 4 banks, conflict-free b128

// workspace layout:
//   [0)       W1t : [n=128][k=256] f16  (64 KB)   transposed for contiguous B-fragments
//   [65536)   W2h : [128] f16           (256 B)
//   [131072)  hsh : relu(hidden_states) f16, [nNodes][128]  (~12.8 MB, L2-resident)

// ---------------------------------------------------------------------------
__global__ void prep_weights(const float* __restrict__ W1,
                             const float* __restrict__ W2,
                             _Float16* __restrict__ W1t,
                             _Float16* __restrict__ W2h) {
  int idx = blockIdx.x * 256 + threadIdx.x;   // 0..32767
  int n = idx >> 8;                           // 0..127
  int k = idx & 255;                          // 0..255
  W1t[n * KDIM + k] = (_Float16)W1[k * HID + n];
  if (idx < HID) W2h[idx] = (_Float16)W2[idx];
}

__global__ void prep_nodes(const float* __restrict__ hs,
                           _Float16* __restrict__ hsh, int total4) {
  int i = blockIdx.x * 256 + threadIdx.x;     // one float4 per thread
  if (i < total4) {
    v4f v = *(const v4f*)(hs + (long long)i * 4);
    v4h o;
#pragma unroll
    for (int j = 0; j < 4; ++j) o[j] = (_Float16)fmaxf(v[j], 0.0f);  // outer relu baked in
    *(v4h*)(hsh + (long long)i * 4) = o;
  }
}

// ---------------------------------------------------------------------------
// 256 threads = 8 waves, 64 edges per block.
// GEMM1: edge_emb[64x256] @ W1[256x128], flat (ks,mt) WMMA sequence with
//        A-fragments prefetched 2 steps ahead (register double-buffer).
// GEMM2: h[64x128] @ W2-broadcast -> logits, waves 0..3.
// ---------------------------------------------------------------------------
__global__ __launch_bounds__(256)
void edge_mlp_kernel(const long long* __restrict__ ei,   // [2, E] int64
                     const _Float16* __restrict__ hsh,   // relu'd f16 node table
                     const _Float16* __restrict__ W1t,
                     const _Float16* __restrict__ W2h,
                     const float* __restrict__ b1,
                     const float* __restrict__ b2,
                     float* __restrict__ out,
                     int nEdges) {
  __shared__ _Float16 sA[MT * LDA];   // 33 KB: concat embeddings (f16, relu'd)
  __shared__ _Float16 sH[MT * LDH];   // 17 KB: post-relu hidden layer (f16)
  __shared__ int sIdx[2 * MT];        // src[64], dst[64]

  const int tid     = threadIdx.x;
  const int lane    = tid & 31;
  const int wave    = tid >> 5;          // N-tile index for GEMM1
  const int l16     = lane & 15;
  const int halfSel = lane >> 4;
  const long long ebase = (long long)blockIdx.x * MT;

  // ---- cache edge indices (int64 -> int) ----
  if (tid < 2 * MT) {
    int e = tid & (MT - 1);
    long long eg = ebase + e;
    long long idx = (tid < MT) ? eg : (long long)nEdges + eg;
    sIdx[tid] = (eg < nEdges) ? (int)ei[idx] : 0;
  }
  __syncthreads();

  // ---- stage sA: pure f16 gather in 16B chunks (no per-element VALU) ----
#pragma unroll
  for (int i = 0; i < 8; ++i) {
    int c  = i * 256 + tid;
    int e  = c >> 5;                 // 32 chunks per edge
    int k8 = (c & 31) * 8;           // 0..248
    int node = (k8 < HID) ? sIdx[e] : sIdx[MT + e];
    v8h d = *(const v8h*)(hsh + (long long)node * HID + (k8 & (HID - 1)));
    *(v8h*)(sA + e * LDA + k8) = d;
  }

  // ---- GEMM1 B-fragments: held in VGPRs, two 16B loads (k = 8h+s, 16+8h+s) ----
  const int n = wave * 16 + l16;
  const _Float16* wcol = W1t + n * KDIM + halfSel * 8;
  v16h B[8];
#pragma unroll
  for (int ks = 0; ks < 8; ++ks)
    B[ks] = __builtin_shufflevector(*(const v8h*)(wcol + ks * 32), *(const v8h*)(wcol + ks * 32 + 16), 0,1,2,3,4,5,6,7,8,9,10,11,12,13,14,15);
  const float b1v = b1[n];

  __syncthreads();

  // ---- GEMM1: flat 32-step (ks outer, mt inner) pipeline, prefetch dist 2 ----
  const _Float16* aBase = sA + l16 * LDA + halfSel * 8;
#define LOAD_A1(mt, ks) ({                                                  \
    const _Float16* _p = aBase + (mt) * (16 * LDA) + (ks) * 32;             \
    v8h _lo = *(const v8h*)(_p);                                            \
    v8h _hi = *(const v8h*)(_p + 16);                                       \
    __builtin_shufflevector(_lo, _hi,                                       \
        0,1,2,3,4,5,6,7,8,9,10,11,12,13,14,15); })

  v8f acc[4] = {{}, {}, {}, {}};
  v16h a0 = LOAD_A1(0, 0);
  v16h a1 = LOAD_A1(1, 0);
#pragma unroll
  for (int t = 0; t < 32; ++t) {
    const int ks = t >> 2;
    const int mt = t & 3;
    const int tn = t + 2;
    v16h an = a1;
    if (tn < 32) an = LOAD_A1(tn & 3, tn >> 2);
    acc[mt] = __builtin_amdgcn_wmma_f32_16x16x32_f16(
                  false, a0, false, B[ks], (short)0, acc[mt], false, false);
    a0 = a1;
    a1 = an;
  }
#undef LOAD_A1

  // bias + relu, write h tiles to sH (f16). C-layout: reg r, lane l ->
  // row m = mt*16 + r + 8*(l/16), col n. b16 stores with immediate offsets.
#pragma unroll
  for (int mt = 0; mt < 4; ++mt) {
    _Float16* hp = sH + (mt * 16 + 8 * halfSel) * LDH + n;
#pragma unroll
    for (int r = 0; r < 8; ++r)
      hp[r * LDH] = (_Float16)fmaxf(acc[mt][r] + b1v, 0.0f);
  }
  __syncthreads();

  // ---- GEMM2: logits = h @ W2 via WMMA with W2 broadcast over N ----
  if (wave < 4) {
    const int mt = wave;
    // B fragment: B[k][n] = W2[k] for all n -> every D column equals the logit
    v16h BW[4];
#pragma unroll
    for (int ks = 0; ks < 4; ++ks)
      BW[ks] = __builtin_shufflevector(*(const v8h*)(W2h + ks * 32 + halfSel * 8), *(const v8h*)(W2h + ks * 32 + halfSel * 8 + 16), 0,1,2,3,4,5,6,7,8,9,10,11,12,13,14,15);

    const _Float16* arow = sH + (mt * 16 + l16) * LDH + halfSel * 8;
    v8f c2 = {};
#pragma unroll
    for (int ks = 0; ks < 4; ++ks) {
      v8h lo = *(const v8h*)(arow + ks * 32);
      v8h hi = *(const v8h*)(arow + ks * 32 + 16);
      v16h a = __builtin_shufflevector(lo, hi,
                 0,1,2,3,4,5,6,7,8,9,10,11,12,13,14,15);
      c2 = __builtin_amdgcn_wmma_f32_16x16x32_f16(
              false, a, false, BW[ks], (short)0, c2, false, false);
    }

    // lanes 0 and 16 hold logits for m = mt*16 + r (+8): two float4 stores
    if (l16 == 0) {
      const float b2v = b2[0];
      long long m0 = ebase + mt * 16 + 8 * halfSel;
      if (m0 + 7 < nEdges) {
        v4f o0 = { c2[0] + b2v, c2[1] + b2v, c2[2] + b2v, c2[3] + b2v };
        v4f o1 = { c2[4] + b2v, c2[5] + b2v, c2[6] + b2v, c2[7] + b2v };
        *(v4f*)(out + m0)     = o0;
        *(v4f*)(out + m0 + 4) = o1;
      } else {
#pragma unroll
        for (int r = 0; r < 8; ++r)
          if (m0 + r < nEdges) out[m0 + r] = c2[r] + b2v;
      }
    }
  }
}

extern "C" void kernel_launch(void* const* d_in, const int* in_sizes, int n_in,
                              void* d_out, int out_size, void* d_ws, size_t ws_size,
                              hipStream_t stream) {
  const float*     hs = (const float*)d_in[0];
  const long long* ei = (const long long*)d_in[1];   // int64 edge_index [2,E]
  const float*     W1 = (const float*)d_in[2];
  const float*     b1 = (const float*)d_in[3];
  const float*     W2 = (const float*)d_in[4];
  const float*     b2 = (const float*)d_in[5];
  float* out = (float*)d_out;

  const int nEdges  = in_sizes[1] / 2;
  const int hsElems = in_sizes[0];        // nNodes * 128

  _Float16* W1t = (_Float16*)d_ws;
  _Float16* W2h = (_Float16*)((char*)d_ws + 65536);
  _Float16* hsh = (_Float16*)((char*)d_ws + 131072);

  prep_weights<<<(KDIM * HID) / 256, 256, 0, stream>>>(W1, W2, W1t, W2h);

  const int total4 = hsElems / 4;
  prep_nodes<<<(total4 + 255) / 256, 256, 0, stream>>>(hs, hsh, total4);

  const int nBlocks = (nEdges + MT - 1) / MT;
  edge_mlp_kernel<<<nBlocks, 256, 0, stream>>>(ei, hsh, W1t, W2h, b1, b2,
                                               out, nEdges);
}
